// TextCnnWithFusion_82360292868101
// MI455X (gfx1250) — matrix operand order repaired and checked by reading, not compiled, not run on hardware
//
#include <hip/hip_runtime.h>
#include <hip/hip_bf16.h>

typedef _Float16 h16;
typedef __attribute__((ext_vector_type(16))) _Float16 v16h;
typedef __attribute__((ext_vector_type(8)))  float    v8f;

#define B_N   2048
#define L     200
#define LP    224     // seq padded to multiple of 32
#define D     100
#define DP    128     // emb padded to multiple of 32
#define DPAD  136     // LDS row stride (halves) -> 4-bank rotation per row
#define NF    50
#define NC    20
#define FEAT  200     // 4 sizes * 50 filters
#define SH_STRIDE 228 // LDS stride for f16 softmax rows
#define WST   648     // LDS stride for conv weight rows (max K=640 + pad)
#define EPSF  1e-13f

// ---- CDNA5 async global->LDS path (ASYNCcnt), with portable fallback ----
#if defined(__AMDGCN__) && __has_builtin(__builtin_amdgcn_global_load_async_to_lds_b128) && __has_builtin(__builtin_amdgcn_s_wait_asynccnt)
#define HAVE_ASYNC_LDS 1
typedef int v4i_t __attribute__((vector_size(16)));
typedef __attribute__((address_space(1))) v4i_t* gptr4;
typedef __attribute__((address_space(3))) v4i_t* lptr4;
#else
#define HAVE_ASYNC_LDS 0
#endif

__device__ __forceinline__ void ldscpy16(h16* dst_lds, const h16* src_glb) {
#if HAVE_ASYNC_LDS
  __builtin_amdgcn_global_load_async_to_lds_b128(
      (gptr4)(void*)(h16*)src_glb, (lptr4)(void*)dst_lds, 0, 0);
#else
  *(uint4*)dst_lds = *(const uint4*)src_glb;
#endif
}

__device__ __forceinline__ void lds_stage_fence() {
#if HAVE_ASYNC_LDS
  __builtin_amdgcn_s_wait_asynccnt(0);
#endif
  __syncthreads();
}

__device__ __forceinline__ v8f wmma16(v16h a, v16h b, v8f c) {
  return __builtin_amdgcn_wmma_f32_16x16x32_f16(false, a, false, b, (short)0, c, false, false);
}

// K index inside a 16x32 f16 A fragment for (lane, half h): contiguous k = 16*(lane/16) + h
__device__ __forceinline__ int a_kmap(int lane, int h) {
  return h + ((lane & 16) ? 16 : 0);
}

// A[m][k] = src[(row0+m)*stride + k0 + k]
__device__ __forceinline__ v16h load_a(const h16* src, int stride, int row0, int k0) {
  int lane = threadIdx.x & 31;
  int m = lane & 15;
  v16h a;
#pragma unroll
  for (int h = 0; h < 16; ++h)
    a[h] = src[(row0 + m) * stride + k0 + a_kmap(lane, h)];
  return a;
}

// B[k][n] = src[(n0+n)*stride + k0 + k]   (transposed operand: lane=n, halves=K)
__device__ __forceinline__ v16h load_bt(const h16* src, int stride, int n0, int k0, int nmax) {
  int lane = threadIdx.x & 31;
  int n = lane & 15;
  int kk = k0 + ((lane & 16) ? 16 : 0);
  v16h b;
#pragma unroll
  for (int h = 0; h < 16; ++h)
    b[h] = (n0 + n < nmax) ? src[(n0 + n) * stride + kk + h] : (h16)0.f;
  return b;
}

// B[k][n] = src[(k0+k)*stride + n0 + n]   (row-major K x N in LDS)
__device__ __forceinline__ v16h load_b(const h16* src, int stride, int k0, int n0) {
  int lane = threadIdx.x & 31;
  int n = lane & 15;
  int kk = k0 + ((lane & 16) ? 16 : 0);
  v16h b;
#pragma unroll
  for (int h = 0; h < 16; ++h)
    b[h] = src[(kk + h) * stride + n0 + n];
  return b;
}

// Conv A fragment: logical K = tap*DP + dim, rows overlap by tap
__device__ __forceinline__ v16h load_a_conv(const h16* src, int row0, int k0) {
  int lane = threadIdx.x & 31;
  int m = lane & 15;
  v16h a;
#pragma unroll
  for (int h = 0; h < 16; ++h) {
    int kk = k0 + a_kmap(lane, h);
    int tap = kk >> 7;
    int k = kk & (DP - 1);
    a[h] = src[(row0 + m + tap) * DPAD + k];
  }
  return a;
}

// ---------------- kernel 1: gather + pad + f16 convert (one block per sample) ----------------
__global__ void gather_embed(const int* __restrict__ x, const float* __restrict__ emb,
                             h16* __restrict__ eh) {
  __shared__ int tok_s[LP];
  int b = blockIdx.x;
  int tid = threadIdx.x;                   // 256 threads
  for (int t = tid; t < LP; t += 256)
    tok_s[t] = (t < L) ? x[b * L + t] : 0;
  __syncthreads();
  h16* dst = eh + (size_t)b * LP * DP;
  for (int i = tid; i < LP * DP / 8; i += 256) {
    int t = i >> 4, kv = (i & 15) * 8;
    const float* erow = emb + (size_t)tok_s[t] * D;
    __builtin_prefetch(erow, 0, 0);
    h16 tmp[8];
#pragma unroll
    for (int q = 0; q < 8; ++q) {
      int k = kv + q;
      float v = (t < L && k < D) ? erow[k] : 0.f;
      tmp[q] = (h16)v;
    }
    *(uint4*)&dst[i * 8] = *(uint4*)tmp;
  }
}

// ---------------- weight/bias repack ----------------
__global__ void repack_w(const float* __restrict__ w, h16* __restrict__ dst, int s) {
  int i = blockIdx.x * blockDim.x + threadIdx.x;
  int total = NF * s * DP;
  if (i >= total) return;
  int f = i / (s * DP);
  int r = i - f * (s * DP);
  int j = r / DP, k = r - j * DP;
  dst[i] = (k < D) ? (h16)w[(f * s + j) * D + k] : (h16)0.f;
}

__global__ void repack_b(const float* b10, const float* b11, const float* b12, const float* b13,
                         const float* b20, const float* b21, const float* b22, const float* b23,
                         float* __restrict__ dst) {
  int i = threadIdx.x;               // 0..511 : [br][sz][64]
  const float* srcs[8] = {b10, b11, b12, b13, b20, b21, b22, b23};
  int br = i >> 8, rem = i & 255, sz = rem >> 6, f = rem & 63;
  dst[i] = (f < NF) ? srcs[br * 4 + sz][f] : 0.f;
}

// ---------------- kernel 2: attention, one block per sample ----------------
__global__ void attn_kernel(const h16* __restrict__ eh, const int* __restrict__ mask,
                            h16* __restrict__ xh) {
  extern __shared__ __align__(16) char smem[];
  h16*   e_s = (h16*)smem;                                            // LP x DPAD
  float* S   = (float*)(smem + LP * DPAD * 2);                        // 16 x LP
  h16*   sh  = (h16*)(smem + LP * DPAD * 2 + 16 * LP * 4);            // 16 x SH_STRIDE
  float* red = (float*)(smem + LP * DPAD * 2 + 16 * LP * 4 + 16 * SH_STRIDE * 2); // 16 x 8
  float* msk = (float*)(smem + LP * DPAD * 2 + 16 * LP * 4 + 16 * SH_STRIDE * 2 + 16 * 8 * 4); // LP

  int b = blockIdx.x;
  int tid = threadIdx.x, wave = tid >> 5, lane = tid & 31;
  const h16* ebase = eh + (size_t)b * LP * DP;

  // stage full sample (224x128 f16) into LDS via async-to-LDS
  for (int i = tid; i < LP * DP / 8; i += 128) {
    int row = i >> 4, cv = i & 15;
    ldscpy16(&e_s[row * DPAD + cv * 8], (const h16*)(((const uint4*)ebase) + i));
  }
  for (int t = tid; t < LP; t += 128)
    msk[t] = (t < L) ? (float)mask[b * L + t] : 0.f;
  lds_stage_fence();

  int r = tid >> 3, sub = tid & 7;
  for (int mt = 0; mt < LP / 16; ++mt) {
    int rg0 = mt * 16;

    // GEMM1: S(16 x LP) = e_rows(16 x DP) * e^T
    for (int nt = wave; nt < LP / 16; nt += 4) {
      v8f acc = {};
#pragma unroll
      for (int kt = 0; kt < DP / 32; ++kt) {
        v16h a  = load_a(e_s, DPAD, rg0, kt * 32);
        v16h bm = load_bt(e_s, DPAD, nt * 16, kt * 32, LP);
        acc = wmma16(a, bm, acc);
      }
      int n = nt * 16 + (lane & 15);
      int mo = (lane & 16) ? 8 : 0;
#pragma unroll
      for (int j = 0; j < 8; ++j)
        S[(j + mo) * LP + n] = acc[j];
    }
    __syncthreads();

    // softmax (diag forced to 0 pre-softmax), zero diag, mask, renormalize (+eps)
    {
      int rg = rg0 + r;
      float mrow = msk[rg];
      float vals[25];
      float lmax = -3.4e38f;
#pragma unroll
      for (int i = 0; i < 25; ++i) {
        int c = sub + 8 * i;                       // c < 200 always
        float s = (c == rg) ? 0.f : S[r * LP + c];
        vals[i] = s;
        lmax = fmaxf(lmax, s);
      }
      red[r * 8 + sub] = lmax;
      __syncthreads();
      float rmax = red[r * 8];
#pragma unroll
      for (int q = 1; q < 8; ++q) rmax = fmaxf(rmax, red[r * 8 + q]);
      __syncthreads();
      float lsum = 0.f;
#pragma unroll
      for (int i = 0; i < 25; ++i) { vals[i] = __expf(vals[i] - rmax); lsum += vals[i]; }
      red[r * 8 + sub] = lsum;
      __syncthreads();
      float tot = 0.f;
#pragma unroll
      for (int q = 0; q < 8; ++q) tot += red[r * 8 + q];
      __syncthreads();
      float inv = 1.f / tot;
      float lsum2 = 0.f;
#pragma unroll
      for (int i = 0; i < 25; ++i) {
        int c = sub + 8 * i;
        float v = vals[i] * inv;
        if (c == rg) v = 0.f;
        v *= mrow * msk[c];
        vals[i] = v;
        lsum2 += v;
      }
      red[r * 8 + sub] = lsum2;
      __syncthreads();
      float tot2 = EPSF;
#pragma unroll
      for (int q = 0; q < 8; ++q) tot2 += red[r * 8 + q];
      float inv2 = 1.f / tot2;
#pragma unroll
      for (int i = 0; i < 25; ++i) {
        int c = sub + 8 * i;
        sh[r * SH_STRIDE + c] = (h16)(vals[i] * inv2);
      }
      for (int c = L + sub; c < LP; c += 8)        // zero padded columns
        sh[r * SH_STRIDE + c] = (h16)0.f;
    }
    __syncthreads();

    // GEMM2: Y(16 x DP) = sh(16 x LP) * e(LP x DP); xh = e + Y (f16, zero-padded)
    for (int nt = wave; nt < DP / 16; nt += 4) {
      v8f acc = {};
#pragma unroll
      for (int kt = 0; kt < LP / 32; ++kt) {
        v16h a  = load_a(sh, SH_STRIDE, 0, kt * 32);
        v16h bm = load_b(e_s, DPAD, kt * 32, nt * 16);
        acc = wmma16(a, bm, acc);
      }
      int col = nt * 16 + (lane & 15);
      int mo = (lane & 16) ? 8 : 0;
#pragma unroll
      for (int j = 0; j < 8; ++j) {
        int t = rg0 + j + mo;
        float outv = acc[j] + (float)e_s[t * DPAD + col];
        h16 hv = (t < L && col < D) ? (h16)outv : (h16)0.f;
        xh[(size_t)b * LP * DP + t * DP + col] = hv;
      }
    }
    __syncthreads();   // sh/S reused next row tile
  }
}

// ---------------- kernel 3: conv-as-GEMM + ReLU + maxpool, one block per (branch, sample) ----------------
__global__ void conv_kernel(const h16* __restrict__ eh, const h16* __restrict__ xh,
                            const h16* __restrict__ wbuf, const float* __restrict__ bbuf,
                            float* __restrict__ pooled) {
  extern __shared__ __align__(16) char smem[];
  h16* in_s = (h16*)smem;                  // LP x DPAD
  h16* w_s  = (h16*)(smem + LP * DPAD * 2);// 64 x WST (rows >= NF masked at frag load)

  const int SZv[4]  = {1, 2, 3, 5};
  const int SOFF[4] = {0, NF, 2 * NF, 3 * NF};
  const int WOFF[4] = {0, NF * DP, NF * 3 * DP, NF * 6 * DP};
  int br = blockIdx.x, b = blockIdx.y;
  int tid = threadIdx.x, wave = tid >> 5, lane = tid & 31;

  const h16* src = (br == 0 ? eh : xh) + (size_t)b * LP * DP;
  for (int i = tid; i < LP * DP / 8; i += 128) {
    int row = i >> 4, cv = i & 15;
    ldscpy16(&in_s[row * DPAD + cv * 8], (const h16*)(((const uint4*)src) + i));
  }

  for (int szi = 0; szi < 4; ++szi) {
    int s = SZv[szi];
    int K = s * DP;
    __syncthreads();                       // previous size's wmma reads of w_s done
    const h16* wsrc = wbuf + br * (NF * 11 * DP) + WOFF[szi];
    for (int i = tid; i < NF * K / 8; i += 128) {
      int f = (i * 8) / K, c = (i * 8) - f * K;
      ldscpy16(&w_s[f * WST + c], (const h16*)(((const uint4*)wsrc) + i));
    }
    lds_stage_fence();                     // covers in_s on first iteration too

    int nt = wave;                         // 4 waves cover filter tiles 0..3 (64 cols)
    int f = nt * 16 + (lane & 15);
    float bias = bbuf[(br * 4 + szi) * 64 + f];
    float runmax[8];
#pragma unroll
    for (int j = 0; j < 8; ++j) runmax[j] = -3.4e38f;

    for (int mtile = 0; mtile < 13; ++mtile) {        // positions 0..207 cover all valid p<=199
      v8f acc = {};
      for (int kt = 0; kt < K / 32; ++kt) {
        v16h a  = load_a_conv(in_s, mtile * 16, kt * 32);
        v16h bm = load_bt(w_s, WST, nt * 16, kt * 32, NF);
        acc = wmma16(a, bm, acc);
      }
      int mo = (lane & 16) ? 8 : 0;
      int pvalid = L - s;                             // last valid conv position
#pragma unroll
      for (int j = 0; j < 8; ++j) {
        int p = mtile * 16 + j + mo;
        if (p <= pvalid) runmax[j] = fmaxf(runmax[j], acc[j] + bias);
      }
    }
    float m = runmax[0];
#pragma unroll
    for (int j = 1; j < 8; ++j) m = fmaxf(m, runmax[j]);
    m = fmaxf(m, __shfl_xor(m, 16));                  // combine M halves (lane n <-> n+16)
    m = fmaxf(m, 0.f);                                // ReLU
    if ((lane < 16) && (f < NF))
      pooled[((size_t)b * 2 + br) * FEAT + SOFF[szi] + f] = m;
  }
}

// ---------------- kernel 4: final FCs + add ----------------
__global__ void fc_kernel(const float* __restrict__ pooled,
                          const float* __restrict__ fw1, const float* __restrict__ fb1,
                          const float* __restrict__ fw2, const float* __restrict__ fb2,
                          float* __restrict__ out) {
  int b = blockIdx.x;
  int c = threadIdx.x;
  if (c >= NC) return;
  const float* p1 = pooled + (size_t)b * 2 * FEAT;
  const float* p2 = p1 + FEAT;
  float acc = fb1[c] + fb2[c];
  for (int k = 0; k < FEAT; ++k)
    acc += p1[k] * fw1[c * FEAT + k] + p2[k] * fw2[c * FEAT + k];
  out[b * NC + c] = acc;
}

extern "C" void kernel_launch(void* const* d_in, const int* in_sizes, int n_in,
                              void* d_out, int out_size, void* d_ws, size_t ws_size,
                              hipStream_t stream) {
  // setup_inputs order: x, x_len, mask, params{emb, cw1[4], cb1[4], cw2[4], cb2[4], fw1, fb1, fw2, fb2}
  const int*   x    = (const int*)d_in[0];
  const int*   mask = (const int*)d_in[2];
  const float* emb  = (const float*)d_in[3];
  const float* cw1[4] = {(const float*)d_in[4], (const float*)d_in[5], (const float*)d_in[6], (const float*)d_in[7]};
  const float* cb1[4] = {(const float*)d_in[8], (const float*)d_in[9], (const float*)d_in[10], (const float*)d_in[11]};
  const float* cw2[4] = {(const float*)d_in[12], (const float*)d_in[13], (const float*)d_in[14], (const float*)d_in[15]};
  const float* cb2[4] = {(const float*)d_in[16], (const float*)d_in[17], (const float*)d_in[18], (const float*)d_in[19]};
  const float* fw1 = (const float*)d_in[20];
  const float* fb1 = (const float*)d_in[21];
  const float* fw2 = (const float*)d_in[22];
  const float* fb2 = (const float*)d_in[23];

  char* ws = (char*)d_ws;
  size_t eh_bytes = (size_t)B_N * LP * DP * sizeof(h16);   // ~112 MB each
  h16* eh   = (h16*)ws;
  h16* xh   = (h16*)(ws + eh_bytes);
  h16* wbuf = (h16*)(ws + 2 * eh_bytes);
  size_t wbuf_bytes = (size_t)2 * NF * 11 * DP * sizeof(h16);
  float* bbuf   = (float*)(ws + 2 * eh_bytes + wbuf_bytes);
  float* pooled = (float*)((char*)bbuf + 512 * sizeof(float));

  gather_embed<<<B_N, 256, 0, stream>>>(x, emb, eh);

  const int SZh[4]  = {1, 2, 3, 5};
  const int WOFFh[4] = {0, NF * DP, NF * 3 * DP, NF * 6 * DP};
  for (int br = 0; br < 2; ++br)
    for (int sz = 0; sz < 4; ++sz) {
      const float* w = (br == 0 ? cw1 : cw2)[sz];
      int total = NF * SZh[sz] * DP;
      repack_w<<<(total + 255) / 256, 256, 0, stream>>>(
          w, wbuf + br * (NF * 11 * DP) + WOFFh[sz], SZh[sz]);
    }
  repack_b<<<1, 512, 0, stream>>>(cb1[0], cb1[1], cb1[2], cb1[3],
                                  cb2[0], cb2[1], cb2[2], cb2[3], bbuf);

  size_t smem2 = (size_t)LP * DPAD * 2 + 16 * LP * 4 + 16 * SH_STRIDE * 2 + 16 * 8 * 4 + LP * 4;
  attn_kernel<<<B_N, 128, smem2, stream>>>(eh, mask, xh);

  size_t smem3 = (size_t)LP * DPAD * 2 + 64 * WST * 2;
  conv_kernel<<<dim3(2, B_N), 128, smem3, stream>>>(eh, xh, wbuf, bbuf, pooled);

  fc_kernel<<<B_N, 32, 0, stream>>>(pooled, fw1, fb1, fw2, fb2, (float*)d_out);
}
